// BiMambaBlock_19662360281191
// MI455X (gfx1250) — matrix operand order repaired and checked by reading, not compiled, not run on hardware
//
#include <hip/hip_runtime.h>
#include <hip/hip_bf16.h>

// ---------------- problem constants ----------------
#define D_MODEL   1024
#define D_STATE   128
#define D_CONV    4
#define HEADDIM   64
#define D_INNER   2048
#define NHEADS    32
#define CONV_DIM  2304            // D_INNER + 2*D_STATE
#define D_IN_PROJ 4384            // 2*D_INNER + 2*D_STATE + NHEADS
#define BATCH     2
#define SEQ       2048
#define M_ROWS    (BATCH*SEQ)     // 4096
#define EPS_F     1e-5f

typedef unsigned short u16;
typedef __attribute__((ext_vector_type(16))) __bf16 v16bf;
typedef __attribute__((ext_vector_type(8)))  __bf16 v8bf;
typedef __attribute__((ext_vector_type(8)))  float  v8f;

// ---------------- bf16 helpers (bit-level, RNE) ----------------
__device__ __forceinline__ u16 f2bf(float f) {
    unsigned u = __float_as_uint(f);
    unsigned r = u + 0x7FFFu + ((u >> 16) & 1u);
    return (u16)(r >> 16);
}
__device__ __forceinline__ float bf2f(u16 h) {
    return __uint_as_float(((unsigned)h) << 16);
}

// ---------------- WMMA fragment loaders (contiguous k order, A and B alike) -------
// A-matrix 16x32 bf16: lane l (m=l&15, h=l>>4) holds A[m, h*16 + 0..15] in
// elements 0..15 (same contiguous k order as the B fragment).
__device__ __forceinline__ v16bf load_a_frag(const u16* A, int lda) {
    int lane = threadIdx.x & 31;
    int m = lane & 15, h = lane >> 4;
    const u16* p = A + (size_t)m * lda + h * 16;
    union { v16bf v; v8bf h8[2]; } u;
    u.h8[0] = *(const v8bf*)(p);
    u.h8[1] = *(const v8bf*)(p + 8);
    return u.v;
}
// B-matrix 32x16 bf16 (B[k,n] = W[n,k], W row-major N x K): lane l
// (n=l&15, h=l>>4) holds B[h*16 + j, n] = W[n, h*16 + j], j = 0..15
// -> one contiguous 32-byte load per lane.
__device__ __forceinline__ v16bf load_b_frag(const u16* W, int ldw) {
    int lane = threadIdx.x & 31;
    int n = lane & 15, h = lane >> 4;
    return *(const v16bf*)(W + (size_t)n * ldw + h * 16);
}

// ---------------- generic WMMA GEMM: C = A(MxK) * W(NxK)^T ---------------
// 128 threads = 4 waves. Wave tile 64x32 (4x2 WMMA accumulators -> 8 WMMAs
// per K-step against 12 b128 loads), block tile 256x32. Compile-time output
// mode: BF16OUT -> bf16 store at column offset; FUSE -> fp32 store with
// fused bias + residual add (final projection). Branch-free epilogue.
template<bool BF16OUT, bool FUSE>
__global__ __launch_bounds__(128) void gemm_bf16_wmma(
    const u16* __restrict__ A, const u16* __restrict__ W, int K,
    float* __restrict__ Cf, u16* __restrict__ Cb, int ldout, int coloff,
    const float* __restrict__ bias, const float* __restrict__ resid)
{
    int wave    = threadIdx.x >> 5;
    int rowbase = blockIdx.x * 256 + wave * 64;
    int colbase = blockIdx.y * 32;

    v8f zero = {0.f,0.f,0.f,0.f,0.f,0.f,0.f,0.f};
    v8f acc[4][2];
    #pragma unroll
    for (int mi = 0; mi < 4; ++mi) { acc[mi][0] = zero; acc[mi][1] = zero; }

    const u16* Ap = A + (size_t)rowbase * K;
    const u16* Wp = W + (size_t)colbase * K;

    #pragma unroll 2
    for (int k0 = 0; k0 < K; k0 += 32) {
        v16bf b0 = load_b_frag(Wp + k0, K);
        v16bf b1 = load_b_frag(Wp + (size_t)16 * K + k0, K);
        v16bf a[4];
        #pragma unroll
        for (int mi = 0; mi < 4; ++mi)
            a[mi] = load_a_frag(Ap + (size_t)(16 * mi) * K + k0, K);
        #pragma unroll
        for (int mi = 0; mi < 4; ++mi) {
            acc[mi][0] = __builtin_amdgcn_wmma_f32_16x16x32_bf16(
                false, a[mi], false, b0, (short)0, acc[mi][0], false, false);
            acc[mi][1] = __builtin_amdgcn_wmma_f32_16x16x32_bf16(
                false, a[mi], false, b1, (short)0, acc[mi][1], false, false);
        }
    }

    int lane = threadIdx.x & 31;
    int nlo = lane & 15, h = lane >> 4;
    #pragma unroll
    for (int mi = 0; mi < 4; ++mi) {
        #pragma unroll
        for (int ni = 0; ni < 2; ++ni) {
            v8f a = acc[mi][ni];
            int col = colbase + ni * 16 + nlo;
            float bv = 0.f;
            if constexpr (FUSE) bv = bias[col];
            #pragma unroll
            for (int r = 0; r < 8; ++r) {
                int row = rowbase + mi * 16 + h * 8 + r;
                size_t oidx = (size_t)row * ldout + coloff + col;
                float v = a[r];
                if constexpr (FUSE) v += bv + resid[oidx];
                if constexpr (BF16OUT) Cb[oidx] = f2bf(v);
                else                   Cf[oidx] = v;
            }
        }
    }
}

// ---------------- fp32 -> bf16 conversion (weights) ----------------------
__global__ __launch_bounds__(256) void cvt_f32_bf16(const float* __restrict__ s,
                                                    u16* __restrict__ d, long n)
{
    long i = (long)blockIdx.x * 256 + threadIdx.x;
    long st = (long)gridDim.x * 256;
    for (; i < n; i += st) d[i] = f2bf(s[i]);
}

// ---------------- RMSNorm of input x -> bf16 ----------------------------
__global__ __launch_bounds__(256) void rmsnorm_in_kernel(
    const float* __restrict__ x, const float* __restrict__ w, u16* __restrict__ out)
{
    __shared__ float red[256];
    int row = blockIdx.x;
    const float* xr = x + (size_t)row * D_MODEL;
    float ss = 0.f;
    for (int c = threadIdx.x; c < D_MODEL; c += 256) { float v = xr[c]; ss += v * v; }
    red[threadIdx.x] = ss; __syncthreads();
    for (int s = 128; s > 0; s >>= 1) {
        if ((int)threadIdx.x < s) red[threadIdx.x] += red[threadIdx.x + s];
        __syncthreads();
    }
    float scale = rsqrtf(red[0] / (float)D_MODEL + EPS_F);
    u16* o = out + (size_t)row * D_MODEL;
    for (int c = threadIdx.x; c < D_MODEL; c += 256) o[c] = f2bf(xr[c] * scale * w[c]);
}

// ---------------- depthwise causal conv + SiLU + dt/dA ------------------
// fwd: out[l] = silu(b + sum_j w[j]*in[l-3+j]); bwd (flip-equivalent in
// original coords): out[l] = silu(b + sum_j w[j]*in[l+3-j]); zero padded.
__global__ __launch_bounds__(256) void conv_dt_kernel(
    const u16* zxf, const u16* zxb,
    const float* cwf, const float* cwb,
    const float* cbf, const float* cbb,
    const float* dtbf, const float* dtbb,
    const float* alogf, const float* alogb,
    float* xcf, float* xcb, float* dtf, float* dtb, float* daf, float* dab)
{
    int dir = blockIdx.y;
    const u16*  zx  = dir ? zxb : zxf;
    const float* cw = dir ? cwb : cwf;
    const float* cb = dir ? cbb : cbf;
    const float* dtbias = dir ? dtbb : dtbf;
    const float* alog   = dir ? alogb : alogf;
    float* xc  = dir ? xcb : xcf;
    float* dtx = dir ? dtb : dtf;
    float* dax = dir ? dab : daf;

    int row = blockIdx.x;            // b*SEQ + l
    int l   = row & (SEQ - 1);

    for (int c = threadIdx.x; c < CONV_DIM; c += 256) {
        float acc = cb[c];
        #pragma unroll
        for (int j = 0; j < D_CONV; ++j) {
            int lsrc = dir ? (l + (D_CONV - 1) - j) : (l - (D_CONV - 1) + j);
            if (lsrc >= 0 && lsrc < SEQ) {
                int srow = row - l + lsrc;
                acc += cw[c * D_CONV + j] *
                       bf2f(zx[(size_t)srow * D_IN_PROJ + D_INNER + c]);
            }
        }
        acc = acc / (1.f + expf(-acc));              // SiLU
        xc[(size_t)row * CONV_DIM + c] = acc;
    }
    if ((int)threadIdx.x < NHEADS) {
        int h = threadIdx.x;
        float v = bf2f(zx[(size_t)row * D_IN_PROJ + D_INNER + CONV_DIM + h]) + dtbias[h];
        float dt = (v > 20.f) ? v : log1pf(expf(v)); // softplus
        float A  = -expf(alog[h]);
        dtx[(size_t)row * NHEADS + h] = dt;
        dax[(size_t)row * NHEADS + h] = expf(dt * A);
    }
}

// ---------------- selective scan (sequential over L) ---------------------
// One block per (b,h,dir). 256 threads; thread owns p = tid>>2 and a 32-wide
// slice of the 128 state columns (state kept in registers). B_t/C_t staged
// into LDS with GLOBAL_LOAD_ASYNC_TO_LDS_B128 (ASYNCcnt), next step's row
// prefetched (global_prefetch_b8); 4-lane partial sums combined with wave32
// shfl_xor.
__global__ __launch_bounds__(256) void scan_kernel(
    const float* xcf, const float* xcb,
    const float* dtf, const float* dtb,
    const float* daf, const float* dab,
    const float* Dskf, const float* Dskb,
    float* yf, float* yb)
{
    int dir = blockIdx.y;
    const float* xc  = dir ? xcb : xcf;
    const float* dtp = dir ? dtb : dtf;
    const float* dap = dir ? dab : daf;
    const float* Dsk = dir ? Dskb : Dskf;
    float* y = dir ? yb : yf;

    int b = blockIdx.x >> 5;         // / NHEADS
    int h = blockIdx.x & 31;
    int tid = threadIdx.x;
    int p = tid >> 2;
    int q = tid & 3;
    int nbase = q * 32;

    __shared__ float sBC[2 * D_STATE];   // [0,128) = B_t, [128,256) = C_t
    unsigned lds_base = (unsigned)(size_t)(&sBC[0]);

    float s[32];
    #pragma unroll
    for (int i = 0; i < 32; ++i) s[i] = 0.f;
    float dsk = Dsk[h];

    for (int step = 0; step < SEQ; ++step) {
        int t = dir ? (SEQ - 1 - step) : step;
        size_t row = (size_t)b * SEQ + t;
        const float* xrow = xc + row * CONV_DIM;

        // async DMA of B_t|C_t (256 floats) into LDS: 64 lanes x b128
        if (tid < 64) {
            unsigned ldsa = lds_base + (unsigned)tid * 16u;
            const float* gp = xrow + D_INNER + tid * 4;
            asm volatile("global_load_async_to_lds_b128 %0, %1, off"
                         :: "v"(ldsa), "v"(gp) : "memory");
        }
        // prefetch next step's B/C row while waiting
        if (step + 1 < SEQ) {
            int tn = dir ? (t - 1) : (t + 1);
            const float* nrow = xc + ((size_t)b * SEQ + tn) * CONV_DIM;
            __builtin_prefetch(nrow + D_INNER + (tid & 63) * 4, 0, 1);
        }
        asm volatile("s_wait_asynccnt 0x0" ::: "memory");
        __syncthreads();

        float dtv = dtp[row * NHEADS + h];
        float dav = dap[row * NHEADS + h];
        float xp  = xrow[h * HEADDIM + p];
        float coef = dtv * xp;
        float acc = 0.f;
        #pragma unroll
        for (int i = 0; i < 32; ++i) {
            s[i] = s[i] * dav + coef * sBC[nbase + i];
            acc += s[i] * sBC[D_STATE + nbase + i];
        }
        acc += __shfl_xor(acc, 1, 32);
        acc += __shfl_xor(acc, 2, 32);
        if (q == 0) y[row * D_INNER + h * HEADDIM + p] = acc + dsk * xp;
        __syncthreads();
    }
}

// ---------------- gating (y * silu(z)) + RMSNorm -> bf16 ------------------
__global__ __launch_bounds__(256) void gate_rmsnorm_kernel(
    const float* yf, const float* yb,
    const u16* zxf, const u16* zxb,
    const float* nwf, const float* nwb,
    u16* ynf, u16* ynb)
{
    int dir = blockIdx.y;
    const float* y  = dir ? yb : yf;
    const u16*  zx  = dir ? zxb : zxf;
    const float* nw = dir ? nwb : nwf;
    u16* yn = dir ? ynb : ynf;

    int row = blockIdx.x;
    __shared__ float red[256];
    __shared__ float buf[D_INNER];
    const float* yr = y + (size_t)row * D_INNER;
    const u16*   zr = zx + (size_t)row * D_IN_PROJ;   // z = first D_INNER cols

    float ss = 0.f;
    for (int c = threadIdx.x; c < D_INNER; c += 256) {
        float z = bf2f(zr[c]);
        float g = yr[c] * (z / (1.f + expf(-z)));
        buf[c] = g; ss += g * g;
    }
    red[threadIdx.x] = ss; __syncthreads();
    for (int s = 128; s > 0; s >>= 1) {
        if ((int)threadIdx.x < s) red[threadIdx.x] += red[threadIdx.x + s];
        __syncthreads();
    }
    float scale = rsqrtf(red[0] / (float)D_INNER + EPS_F);
    u16* o = yn + (size_t)row * D_INNER;
    for (int c = threadIdx.x; c < D_INNER; c += 256) o[c] = f2bf(buf[c] * scale * nw[c]);
}

// ---------------- host side ----------------------------------------------
extern "C" void kernel_launch(void* const* d_in, const int* in_sizes, int n_in,
                              void* d_out, int out_size, void* d_ws, size_t ws_size,
                              hipStream_t stream)
{
    const float* x        = (const float*)d_in[0];
    const float* norm_w   = (const float*)d_in[1];
    const float* fW_in    = (const float*)d_in[2];
    const float* fconv_w  = (const float*)d_in[3];
    const float* fconv_b  = (const float*)d_in[4];
    const float* fdt_bias = (const float*)d_in[5];
    const float* fA_log   = (const float*)d_in[6];
    const float* fD       = (const float*)d_in[7];
    const float* fnorm    = (const float*)d_in[8];
    const float* fW_out   = (const float*)d_in[9];
    const float* bW_in    = (const float*)d_in[10];
    const float* bconv_w  = (const float*)d_in[11];
    const float* bconv_b  = (const float*)d_in[12];
    const float* bdt_bias = (const float*)d_in[13];
    const float* bA_log   = (const float*)d_in[14];
    const float* bD       = (const float*)d_in[15];
    const float* bnorm    = (const float*)d_in[16];
    const float* bW_out   = (const float*)d_in[17];
    const float* projW    = (const float*)d_in[18];
    const float* projb    = (const float*)d_in[19];
    float* out = (float*)d_out;

    // workspace carve (256B aligned)
    char* ws = (char*)d_ws;
    size_t off = 0;
    auto carve = [&](size_t bytes) -> void* {
        void* p = ws + off;
        off += (bytes + 255) & ~(size_t)255;
        return p;
    };
    u16*  xn      = (u16*)carve((size_t)M_ROWS * D_MODEL * 2);
    u16*  fWin_b  = (u16*)carve((size_t)D_IN_PROJ * D_MODEL * 2);
    u16*  bWin_b  = (u16*)carve((size_t)D_IN_PROJ * D_MODEL * 2);
    u16*  fWout_b = (u16*)carve((size_t)D_MODEL * D_INNER * 2);
    u16*  bWout_b = (u16*)carve((size_t)D_MODEL * D_INNER * 2);
    u16*  projW_b = (u16*)carve((size_t)D_MODEL * D_INNER * 2);
    u16*  zx_f    = (u16*)carve((size_t)M_ROWS * D_IN_PROJ * 2);
    u16*  zx_b    = (u16*)carve((size_t)M_ROWS * D_IN_PROJ * 2);
    float* xc_f   = (float*)carve((size_t)M_ROWS * CONV_DIM * 4);
    float* xc_b   = (float*)carve((size_t)M_ROWS * CONV_DIM * 4);
    float* dt_f   = (float*)carve((size_t)M_ROWS * NHEADS * 4);
    float* dt_b   = (float*)carve((size_t)M_ROWS * NHEADS * 4);
    float* da_f   = (float*)carve((size_t)M_ROWS * NHEADS * 4);
    float* da_b   = (float*)carve((size_t)M_ROWS * NHEADS * 4);
    float* y_f    = (float*)carve((size_t)M_ROWS * D_INNER * 4);
    float* y_b    = (float*)carve((size_t)M_ROWS * D_INNER * 4);
    u16*  yn_f    = (u16*)carve((size_t)M_ROWS * D_INNER * 2);
    u16*  yn_b    = (u16*)carve((size_t)M_ROWS * D_INNER * 2);
    u16*  ycat    = (u16*)carve((size_t)M_ROWS * (2 * D_MODEL) * 2);
    (void)ws_size; (void)in_sizes; (void)n_in; (void)out_size;

    // 1) weight conversions to bf16
    cvt_f32_bf16<<<512, 256, 0, stream>>>(fW_in,  fWin_b,  (long)D_IN_PROJ * D_MODEL);
    cvt_f32_bf16<<<512, 256, 0, stream>>>(bW_in,  bWin_b,  (long)D_IN_PROJ * D_MODEL);
    cvt_f32_bf16<<<512, 256, 0, stream>>>(fW_out, fWout_b, (long)D_MODEL * D_INNER);
    cvt_f32_bf16<<<512, 256, 0, stream>>>(bW_out, bWout_b, (long)D_MODEL * D_INNER);
    cvt_f32_bf16<<<512, 256, 0, stream>>>(projW,  projW_b, (long)D_MODEL * D_INNER);

    // 2) RMSNorm(x) -> bf16 (shared by both directions)
    rmsnorm_in_kernel<<<M_ROWS, 256, 0, stream>>>(x, norm_w, xn);

    // 3) in_proj GEMMs (bf16 output)
    dim3 g_in(M_ROWS / 256, D_IN_PROJ / 32);
    gemm_bf16_wmma<true,false><<<g_in, 128, 0, stream>>>(xn, fWin_b, D_MODEL,
        nullptr, zx_f, D_IN_PROJ, 0, nullptr, nullptr);
    gemm_bf16_wmma<true,false><<<g_in, 128, 0, stream>>>(xn, bWin_b, D_MODEL,
        nullptr, zx_b, D_IN_PROJ, 0, nullptr, nullptr);

    // 4) depthwise conv + SiLU + dt/dA (both directions)
    conv_dt_kernel<<<dim3(M_ROWS, 2), 256, 0, stream>>>(
        zx_f, zx_b, fconv_w, bconv_w, fconv_b, bconv_b,
        fdt_bias, bdt_bias, fA_log, bA_log,
        xc_f, xc_b, dt_f, dt_b, da_f, da_b);

    // 5) selective scan (both directions)
    scan_kernel<<<dim3(BATCH * NHEADS, 2), 256, 0, stream>>>(
        xc_f, xc_b, dt_f, dt_b, da_f, da_b, fD, bD, y_f, y_b);

    // 6) gate + RMSNorm -> bf16
    gate_rmsnorm_kernel<<<dim3(M_ROWS, 2), 256, 0, stream>>>(
        y_f, y_b, zx_f, zx_b, fnorm, bnorm, yn_f, yn_b);

    // 7) out_proj GEMMs -> concatenated bf16 (ycat: [y_fwd | y_bwd])
    dim3 g_out(M_ROWS / 256, D_MODEL / 32);
    gemm_bf16_wmma<true,false><<<g_out, 128, 0, stream>>>(yn_f, fWout_b, D_INNER,
        nullptr, ycat, 2 * D_MODEL, 0,       nullptr, nullptr);
    gemm_bf16_wmma<true,false><<<g_out, 128, 0, stream>>>(yn_b, bWout_b, D_INNER,
        nullptr, ycat, 2 * D_MODEL, D_MODEL, nullptr, nullptr);

    // 8) final projection + bias + residual -> fp32 d_out
    gemm_bf16_wmma<false,true><<<g_out, 128, 0, stream>>>(ycat, projW_b, 2 * D_MODEL,
        out, nullptr, D_MODEL, 0, projb, x);
}
